// OrientedRCNNHead_65859028517276
// MI455X (gfx1250) — matrix operand order repaired and checked by reading, not compiled, not run on hardware
//
#include <hip/hip_runtime.h>
#include <stdint.h>

// Problem constants (match the reference)
#define Bb    2
#define Nroi  1000
#define Cch   256
#define Hf    256
#define Wf    256
#define PHb   7
#define PWb   7
#define Dfeat (Cch*PHb*PWb)   // 12544
#define HID   1024
#define NCLS  11
#define NREG  5
#define Mrows (Bb*Nroi)       // 2000

typedef __attribute__((ext_vector_type(16))) __bf16 v16bf;
typedef __attribute__((ext_vector_type(8)))  float  v8f;

union Frag16 { v16bf v; uint4 q[2]; };

__device__ __forceinline__ unsigned short f2bf(float f) {
  unsigned u = __float_as_uint(f);
  u += 0x7fffu + ((u >> 16) & 1u);           // round-to-nearest-even
  return (unsigned short)(u >> 16);
}
__device__ __forceinline__ float bf2f(unsigned short h) {
  return __uint_as_float(((unsigned)h) << 16);
}

// LDS byte offset of a __shared__ object (addrspace(3) value == LDS offset)
typedef __attribute__((address_space(3))) unsigned short lds_us;
__device__ __forceinline__ unsigned lds_off(const unsigned short* p) {
  return (unsigned)(uintptr_t)(lds_us*)p;
}

// CDNA5 async copy: 16B global -> LDS, tracked by ASYNCcnt (GVS addressing:
// 64-bit SGPR base + 32-bit unsigned VGPR byte offset; vdst = LDS byte addr)
#define ASYNC_CP16(ldsb, goffb, base)                                   \
  asm volatile("global_load_async_to_lds_b128 %0, %1, %2"               \
               :: "v"(ldsb), "v"(goffb), "s"(base) : "memory")
#define WAIT_ASYNC_12() asm volatile("s_wait_asynccnt 0xc" ::: "memory")
#define WAIT_ASYNC_0()  asm volatile("s_wait_asynccnt 0x0" ::: "memory")

// ---------------------------------------------------------------------------
// Kernel 1: rotated RoI-align, writes split-bf16 activations X = [2000][12544]
// ---------------------------------------------------------------------------
__global__ __launch_bounds__(256) void roi_align_kernel(
    const float* __restrict__ feat, const float* __restrict__ rois,
    unsigned short* __restrict__ Xhi, unsigned short* __restrict__ Xlo) {
  int idx = blockIdx.x * 256 + threadIdx.x;
  const int total = Mrows * Dfeat;
  if (idx >= total) return;
  int row = idx / Dfeat;
  int col = idx - row * Dfeat;
  int c   = col / (PHb*PWb);
  int p   = col - c * (PHb*PWb);
  int ph  = p / PWb;
  int pw  = p - ph * PWb;
  int b   = row / Nroi;

  const float* r = rois + (size_t)row * 5;
  float cx = r[0], cy = r[1], w = r[2], h = r[3], th = r[4];
  float bin_h = h * (1.0f/PHb), bin_w = w * (1.0f/PWb);
  float cs = cosf(th), sn = sinf(th);
  const float* f = feat + ((size_t)b * Cch + c) * (size_t)(Hf*Wf);

  float sum = 0.f;
#pragma unroll
  for (int iy = 0; iy < 2; ++iy) {
    float yy = -h*0.5f + ((float)ph + ((float)iy + 0.5f)*0.5f) * bin_h;
#pragma unroll
    for (int ix = 0; ix < 2; ++ix) {
      float xx = -w*0.5f + ((float)pw + ((float)ix + 0.5f)*0.5f) * bin_w;
      float x = cx + xx*cs - yy*sn;
      float y = cy + xx*sn + yy*cs;
      if (y > -1.0f && y < (float)Hf && x > -1.0f && x < (float)Wf) {
        float yc = fminf(fmaxf(y, 0.f), (float)(Hf-1));
        float xc = fminf(fmaxf(x, 0.f), (float)(Wf-1));
        float y0 = floorf(yc), x0 = floorf(xc);
        int y0i = (int)y0, x0i = (int)x0;
        int y1i = min(y0i + 1, Hf-1), x1i = min(x0i + 1, Wf-1);
        float ly = yc - y0, lx = xc - x0;
        float hy = 1.f - ly, hx = 1.f - lx;
        float v00 = f[y0i*Wf + x0i], v01 = f[y0i*Wf + x1i];
        float v10 = f[y1i*Wf + x0i], v11 = f[y1i*Wf + x1i];
        sum += hy*hx*v00 + hy*lx*v01 + ly*hx*v10 + ly*lx*v11;
      }
    }
  }
  sum *= 0.25f;
  unsigned short hi = f2bf(sum);
  Xhi[idx] = hi;
  Xlo[idx] = f2bf(sum - bf2f(hi));
}

// ---------------------------------------------------------------------------
// Kernel 2: transpose + split weights: W[K][N] fp32 -> WhiT/WloT[N][K] bf16
// ---------------------------------------------------------------------------
__global__ __launch_bounds__(256) void split_tr_kernel(
    const float* __restrict__ W, unsigned short* __restrict__ WhiT,
    unsigned short* __restrict__ WloT, int K, int N) {
  __shared__ float tile[32][33];
  int kb = blockIdx.x * 32, nb = blockIdx.y * 32;
  int tx = threadIdx.x, ty = threadIdx.y;
  for (int i = ty; i < 32; i += 8)
    tile[i][tx] = W[(size_t)(kb + i) * N + (nb + tx)];
  __syncthreads();
  for (int i = ty; i < 32; i += 8) {
    float v = tile[tx][i];                 // = W[kb+tx][nb+i]
    size_t o = (size_t)(nb + i) * K + (kb + tx);
    unsigned short hi = f2bf(v);
    WhiT[o] = hi;
    WloT[o] = f2bf(v - bf2f(hi));
  }
}

// ---------------------------------------------------------------------------
// Kernel 3: split-bf16 GEMM, fp32 accumulation:
//   C = relu( Ahi@Bhi + Ahi@Blo + Alo@Bhi + bias )
// WG tile 64(M) x 128(N); 8 waves (2x4); wave = 32x32 via 2x2 WMMA frags.
// K staged in chunks of 64 through double-buffered LDS filled by CDNA5
// global_load_async_to_lds_b128 (ASYNCcnt pipeline: issue k+1, drain k, go).
// ---------------------------------------------------------------------------
#define TM  64
#define TN  128
#define TK2 64
#define LDT 72    // halfwords per tile row (144B: 16B aligned, conflict-free)

template<int WRITE_BF16>
__global__ __launch_bounds__(256) void gemm_bf16x3_kernel(
    const unsigned short* __restrict__ Ahi, const unsigned short* __restrict__ Alo,
    const unsigned short* __restrict__ BhiT, const unsigned short* __restrict__ BloT,
    const float* __restrict__ bias,
    float* __restrict__ Cf, unsigned short* __restrict__ Chi, unsigned short* __restrict__ Clo,
    int M, int N, int K) {
  __shared__ alignas(16) unsigned short sAhi[2][TM*LDT], sAlo[2][TM*LDT];
  __shared__ alignas(16) unsigned short sBhi[2][TN*LDT], sBlo[2][TN*LDT];

  const int tid   = threadIdx.x;
  const int lane  = tid & 31;
  const int wid   = tid >> 5;        // 0..7
  const int waveM = wid & 1;         // 2 blocks of 32 rows
  const int waveN = wid >> 1;        // 4 blocks of 32 cols
  const int mBase = blockIdx.y * TM;
  const int nBase = blockIdx.x * TN;

  const int lr = tid >> 2;           // 0..63 : tile row staged by this thread
  const int lc = (tid & 3) * 8;      // halfword chunk base 0,8,16,24

  // Clamped A row: rows >= M duplicate row M-1 (their outputs are never
  // stored) so every async address is in-bounds and EXEC stays uniform.
  const int      grA   = min(mBase + lr, M - 1);
  const unsigned aoffb = (unsigned)((unsigned)grA * (unsigned)K) * 2u;  // bytes
  const unsigned b0offb = (unsigned)((unsigned)(nBase + lr)      * (unsigned)K) * 2u;
  const unsigned b1offb = (unsigned)((unsigned)(nBase + lr + 64) * (unsigned)K) * 2u;

  v8f acc[2][2] = {};

  const int nk = K / TK2;

  // issue one K-stage (12 x 16B async copies per thread) into buffer bf
  auto issue_stage = [&](int k0, int bf) {
    const unsigned kb = (unsigned)k0 * 2u;            // byte offset of stage
    const unsigned c0 = (unsigned)lc * 2u;            // chunk 0 (bytes)
    const unsigned c1 = c0 + 64u;                     // chunk 1 (+32 halfwords)
    unsigned lA  = lds_off(&sAhi[bf][lr*LDT]);
    unsigned lAl = lds_off(&sAlo[bf][lr*LDT]);
    unsigned lB0 = lds_off(&sBhi[bf][lr*LDT]);
    unsigned lB0l= lds_off(&sBlo[bf][lr*LDT]);
    unsigned lB1 = lds_off(&sBhi[bf][(lr+64)*LDT]);
    unsigned lB1l= lds_off(&sBlo[bf][(lr+64)*LDT]);
    ASYNC_CP16(lA  + c0, aoffb  + kb + c0, Ahi);
    ASYNC_CP16(lA  + c1, aoffb  + kb + c1, Ahi);
    ASYNC_CP16(lAl + c0, aoffb  + kb + c0, Alo);
    ASYNC_CP16(lAl + c1, aoffb  + kb + c1, Alo);
    ASYNC_CP16(lB0 + c0, b0offb + kb + c0, BhiT);
    ASYNC_CP16(lB0 + c1, b0offb + kb + c1, BhiT);
    ASYNC_CP16(lB1 + c0, b1offb + kb + c0, BhiT);
    ASYNC_CP16(lB1 + c1, b1offb + kb + c1, BhiT);
    ASYNC_CP16(lB0l+ c0, b0offb + kb + c0, BloT);
    ASYNC_CP16(lB0l+ c1, b0offb + kb + c1, BloT);
    ASYNC_CP16(lB1l+ c0, b1offb + kb + c0, BloT);
    ASYNC_CP16(lB1l+ c1, b1offb + kb + c1, BloT);
  };

  issue_stage(0, 0);

  const int mr  = lane & 15;
  const int sel = lane >> 4;

  for (int kt = 0; kt < nk; ++kt) {
    const int cur = kt & 1, nxt = cur ^ 1;
    if (kt + 1 < nk) {                 // prefetch next stage, drain current
      issue_stage((kt + 1) * TK2, nxt);
      WAIT_ASYNC_12();
    } else {
      WAIT_ASYNC_0();
    }
    __syncthreads();                   // all waves' stage `cur` landed in LDS

    const unsigned short* pAh = sAhi[cur];
    const unsigned short* pAl = sAlo[cur];
    const unsigned short* pBh = sBhi[cur];
    const unsigned short* pBl = sBlo[cur];

#pragma unroll
    for (int kk = 0; kk < 2; ++kk) {   // two 16x16x32 chunks per stage
      Frag16 a_hi[2], a_lo[2], b_hi[2], b_lo[2];
#pragma unroll
      for (int mi = 0; mi < 2; ++mi) {
        int base = (waveM*32 + mi*16 + mr)*LDT + kk*32 + sel*8;
        a_hi[mi].q[0] = *(const uint4*)(pAh + base);
        a_hi[mi].q[1] = *(const uint4*)(pAh + base + 16);
        a_lo[mi].q[0] = *(const uint4*)(pAl + base);
        a_lo[mi].q[1] = *(const uint4*)(pAl + base + 16);
      }
#pragma unroll
      for (int ni = 0; ni < 2; ++ni) {
        int base = (waveN*32 + ni*16 + mr)*LDT + kk*32 + sel*8;
        b_hi[ni].q[0] = *(const uint4*)(pBh + base);
        b_hi[ni].q[1] = *(const uint4*)(pBh + base + 16);
        b_lo[ni].q[0] = *(const uint4*)(pBl + base);
        b_lo[ni].q[1] = *(const uint4*)(pBl + base + 16);
      }
#pragma unroll
      for (int mi = 0; mi < 2; ++mi)
#pragma unroll
        for (int ni = 0; ni < 2; ++ni) {
          acc[mi][ni] = __builtin_amdgcn_wmma_f32_16x16x32_bf16(
              false, a_hi[mi].v, false, b_hi[ni].v, (short)0, acc[mi][ni], false, false);
          acc[mi][ni] = __builtin_amdgcn_wmma_f32_16x16x32_bf16(
              false, a_hi[mi].v, false, b_lo[ni].v, (short)0, acc[mi][ni], false, false);
          acc[mi][ni] = __builtin_amdgcn_wmma_f32_16x16x32_bf16(
              false, a_lo[mi].v, false, b_hi[ni].v, (short)0, acc[mi][ni], false, false);
        }
    }
    __syncthreads();                   // stage `cur` free for next prefetch
  }

  // ---- epilogue: bias + ReLU; fp32 out or re-split bf16 hi/lo out ----
  const int rAdd = (lane < 16) ? 0 : 8;   // C/D frag: VGPR r -> M = r or r+8
#pragma unroll
  for (int mi = 0; mi < 2; ++mi)
#pragma unroll
    for (int ni = 0; ni < 2; ++ni) {
      int colI = nBase + waveN*32 + ni*16 + mr;
      float bv = bias[colI];
#pragma unroll
      for (int r = 0; r < 8; ++r) {
        int row = mBase + waveM*32 + mi*16 + r + rAdd;
        if (row < M) {
          float v = fmaxf(acc[mi][ni][r] + bv, 0.0f);
          size_t o = (size_t)row * N + colI;
          if (WRITE_BF16) {
            unsigned short hv = f2bf(v);
            Chi[o] = hv;
            Clo[o] = f2bf(v - bf2f(hv));
          } else {
            Cf[o] = v;
          }
        }
      }
    }
}

// ---------------------------------------------------------------------------
// Kernel 4: heads: cls = H2@wc+bc (N=11), reg = H2@wr+br (N=5); tiny fp32 work
// ---------------------------------------------------------------------------
__global__ __launch_bounds__(128) void heads_kernel(
    const float* __restrict__ H2, const float* __restrict__ wc, const float* __restrict__ bc,
    const float* __restrict__ wr, const float* __restrict__ br, float* __restrict__ out) {
  __shared__ float x[HID];
  __shared__ float part[128];
  const int row = blockIdx.x;
  const int tid = threadIdx.x;
  for (int i = tid; i < HID; i += 128) x[i] = H2[(size_t)row*HID + i];
  __syncthreads();
  const int o = tid >> 3, p = tid & 7;
  float s = 0.f;
  if (o < NCLS) {
    for (int k = p; k < HID; k += 8) s += x[k] * wc[(size_t)k*NCLS + o];
  } else {
    int j = o - NCLS;
    for (int k = p; k < HID; k += 8) s += x[k] * wr[(size_t)k*NREG + j];
  }
  part[tid] = s;
  __syncthreads();
  if (tid < NCLS + NREG) {
    float t = 0.f;
#pragma unroll
    for (int q = 0; q < 8; ++q) t += part[tid*8 + q];
    if (tid < NCLS) out[(size_t)row*NCLS + tid] = t + bc[tid];
    else out[(size_t)Mrows*NCLS + (size_t)row*NREG + (tid - NCLS)] = t + br[tid - NCLS];
  }
}

// ---------------------------------------------------------------------------
extern "C" void kernel_launch(void* const* d_in, const int* in_sizes, int n_in,
                              void* d_out, int out_size, void* d_ws, size_t ws_size,
                              hipStream_t stream) {
  (void)in_sizes; (void)n_in; (void)out_size; (void)ws_size;
  const float* feat = (const float*)d_in[0];
  const float* rois = (const float*)d_in[1];
  const float* w1   = (const float*)d_in[2];
  const float* b1   = (const float*)d_in[3];
  const float* w2   = (const float*)d_in[4];
  const float* b2   = (const float*)d_in[5];
  const float* wc   = (const float*)d_in[6];
  const float* bc   = (const float*)d_in[7];
  const float* wr   = (const float*)d_in[8];
  const float* br   = (const float*)d_in[9];
  float* out = (float*)d_out;

  char* p = (char*)d_ws;
  auto take = [&](size_t bytes) -> char* {
    char* r = p; p += (bytes + 255) & ~(size_t)255; return r;
  };
  unsigned short* Xhi   = (unsigned short*)take((size_t)Mrows*Dfeat*2);
  unsigned short* Xlo   = (unsigned short*)take((size_t)Mrows*Dfeat*2);
  unsigned short* W1hiT = (unsigned short*)take((size_t)Dfeat*HID*2);
  unsigned short* W1loT = (unsigned short*)take((size_t)Dfeat*HID*2);
  unsigned short* W2hiT = (unsigned short*)take((size_t)HID*HID*2);
  unsigned short* W2loT = (unsigned short*)take((size_t)HID*HID*2);
  unsigned short* H1hi  = (unsigned short*)take((size_t)Mrows*HID*2);
  unsigned short* H1lo  = (unsigned short*)take((size_t)Mrows*HID*2);
  float*          H2    = (float*)take((size_t)Mrows*HID*4);

  // 1) RoI-align -> split-bf16 activations
  {
    int total = Mrows * Dfeat;
    roi_align_kernel<<<(total + 255) / 256, 256, 0, stream>>>(feat, rois, Xhi, Xlo);
  }
  // 2) weight transpose + split
  split_tr_kernel<<<dim3(Dfeat/32, HID/32), dim3(32,8), 0, stream>>>(w1, W1hiT, W1loT, Dfeat, HID);
  split_tr_kernel<<<dim3(HID/32,   HID/32), dim3(32,8), 0, stream>>>(w2, W2hiT, W2loT, HID,   HID);
  // 3) GEMM1: X(2000x12544) @ W1(12544x1024) + b1, ReLU -> H1 (split bf16)
  gemm_bf16x3_kernel<1><<<dim3(HID/TN, (Mrows + TM - 1)/TM), 256, 0, stream>>>(
      Xhi, Xlo, W1hiT, W1loT, b1, nullptr, H1hi, H1lo, Mrows, HID, Dfeat);
  // 4) GEMM2: H1 @ W2 + b2, ReLU -> H2 (fp32)
  gemm_bf16x3_kernel<0><<<dim3(HID/TN, (Mrows + TM - 1)/TM), 256, 0, stream>>>(
      H1hi, H1lo, W2hiT, W2loT, b2, H2, nullptr, nullptr, Mrows, HID, HID);
  // 5) heads -> d_out (cls flat then reg flat)
  heads_kernel<<<Mrows, 128, 0, stream>>>(H2, wc, bc, wr, br, out);
}
